// BiLSTM_CRF_79491254714706
// MI455X (gfx1250) — matrix operand order repaired and checked by reading, not compiled, not run on hardware
//
#include <hip/hip_runtime.h>
#include <hip/hip_bf16.h>
#include <math.h>

// ---------------------------------------------------------------------------
// BiLSTM-CRF forward for MI455X (gfx1250, wave32, WMMA).
// B=64, T=512, V=30000, E=512, H=256, L=2, NT=20.
// ---------------------------------------------------------------------------

typedef __attribute__((ext_vector_type(8)))  __bf16 v8bf;
typedef __attribute__((ext_vector_type(16))) __bf16 v16bf;
typedef __attribute__((ext_vector_type(8)))  float  v8f;

static __device__ __forceinline__ v16bf cat8(v8bf lo, v8bf hi) {
  return __builtin_shufflevector(lo, hi, 0,1,2,3,4,5,6,7,8,9,10,11,12,13,14,15);
}

// A-matrix 16x32 bf16 tile (row-major source, leading dim `ld` elements).
// ISA 7.12.2: lanes 0-15 -> M=lane, K=k0+[0..7] then k0+[16..23];
//             lanes 16-31 -> M=lane-16, K=k0+[8..15] then k0+[24..31].
static __device__ __forceinline__ v16bf load_a16x32(const __bf16* tile, int ld, int lane) {
  const __bf16* p = tile + (lane & 15) * ld + ((lane >> 4) << 3);
  v8bf lo = *(const v8bf*)p;
  v8bf hi = *(const v8bf*)(p + 16);
  return cat8(lo, hi);
}

// B-matrix 32x16 bf16 tile from W stored [N][K] row-major (so B col n = W row n).
// lanes 0-15 -> N=lane, K=k0+[0..7] then k0+[16..23]; lanes 16-31 -> N=lane-16, K=k0+[8..15] then k0+[24..31].
static __device__ __forceinline__ v16bf load_b32x16(const __bf16* wtile, int ld, int lane) {
  const __bf16* p = wtile + (lane & 15) * ld + ((lane >> 4) << 3);
  v8bf lo = *(const v8bf*)p;
  v8bf hi = *(const v8bf*)(p + 16);
  return cat8(lo, hi);
}

static __device__ __forceinline__ v8f wmma_bf16(v16bf a, v16bf b, v8f c) {
  return __builtin_amdgcn_wmma_f32_16x16x32_bf16(false, a, false, b, (short)0, c, false, false);
}

static __device__ __forceinline__ float sigf(float x) { return 1.0f / (1.0f + __expf(-x)); }

// ---------------------------------------------------------------------------
__global__ void __launch_bounds__(32) zero_kernel(float* out) {
  if (threadIdx.x == 0) out[0] = 0.0f;
}

__global__ void __launch_bounds__(256) f32_to_bf16_kernel(const float* __restrict__ in,
                                                          __bf16* __restrict__ out, int n) {
  int i = blockIdx.x * 256 + threadIdx.x;
  if (i < n) out[i] = (__bf16)in[i];
}

// x_bf16[b*T+t][e] = bf16(embed[ids[b*T+t]][e]); 8 elems per thread.
__global__ void __launch_bounds__(256) gather_embed_kernel(const int* __restrict__ ids,
                                                           const float* __restrict__ embed,
                                                           __bf16* __restrict__ x) {
  size_t idx = (size_t)blockIdx.x * 256 + threadIdx.x;   // one 8-elem chunk
  int m  = (int)(idx >> 6);                              // 512/8 = 64 chunks per row
  int e0 = (int)(idx & 63) << 3;
  int id = ids[m];
  const float* src = embed + (size_t)id * 512 + e0;
  v8bf o;
#pragma unroll
  for (int i = 0; i < 8; ++i) o[i] = (__bf16)src[i];
  *(v8bf*)(x + (size_t)m * 512 + e0) = o;
}

// ---------------------------------------------------------------------------
// xg[m][n] = sum_k A[m][k]*Wih[layer][dir][gi][k] + b_ih + b_hh
// M = 32768, N = 2048 (dir*1024 + gate*256 + h), K = 512.
// Block = 8 waves; wave computes a 16(M) x 64(N) strip (4 WMMA tiles).
// ---------------------------------------------------------------------------
__global__ void __launch_bounds__(256) xg_gemm_kernel(const __bf16* __restrict__ A,
                                                      const __bf16* __restrict__ wih,
                                                      const float* __restrict__ bih,
                                                      const float* __restrict__ bhh,
                                                      float* __restrict__ xg, int layer) {
  const int tid = threadIdx.x, wave = tid >> 5, lane = tid & 31;
  const int m0 = blockIdx.x * 128 + wave * 16;
  const int n0 = blockIdx.y * 64;

  v8f acc[4] = {};
  const __bf16* wbase[4];
  float biasv[4];
#pragma unroll
  for (int j = 0; j < 4; ++j) {
    int ncol = n0 + j * 16;                 // tile never crosses the dir boundary
    int d = ncol >> 10, gi0 = ncol & 1023;
    wbase[j] = wih + ((size_t)((layer * 2 + d) * 1024 + gi0)) * 512;
    int bidx = (layer * 2 + d) * 1024 + gi0 + (lane & 15);
    biasv[j] = bih[bidx] + bhh[bidx];
  }
  const __bf16* abase = A + (size_t)m0 * 512;

#pragma unroll 4
  for (int kt = 0; kt < 16; ++kt) {
    const int k0 = kt * 32;
    v16bf a = load_a16x32(abase + k0, 512, lane);
#pragma unroll
    for (int j = 0; j < 4; ++j) {
      v16bf b = load_b32x16(wbase[j] + k0, 512, lane);
      acc[j] = wmma_bf16(a, b, acc[j]);
    }
  }
#pragma unroll
  for (int j = 0; j < 4; ++j) {
    int col = n0 + j * 16 + (lane & 15);
#pragma unroll
    for (int r = 0; r < 8; ++r) {
      int row = m0 + r + ((lane >> 4) << 3);
      xg[(size_t)row * 2048 + col] = acc[j][r] + biasv[j];
    }
  }
}

// ---------------------------------------------------------------------------
// Sequential LSTM scan, one workgroup per direction (blockIdx.x = dir).
// 32 waves; per step each wave computes 2 (m-tile,h-tile) groups x 4 gate
// tiles via WMMA with C initialized from xg, gates fused in registers,
// h kept in LDS (bf16), c persistent in VGPRs, w_hh served from L2.
// ---------------------------------------------------------------------------
__global__ void __launch_bounds__(1024) lstm_scan_kernel(const float* __restrict__ xg,
                                                         const __bf16* __restrict__ whh,
                                                         const int* __restrict__ lengths,
                                                         __bf16* __restrict__ out, int layer) {
  const int dir  = blockIdx.x;
  const int tid  = threadIdx.x;
  const int wave = tid >> 5, lane = tid & 31;

  __shared__ __bf16 h_lds[64][256];
  __shared__ int lens[64];
  for (int i = tid; i < 64 * 256; i += 1024) ((__bf16*)h_lds)[i] = (__bf16)0.0f;
  if (tid < 64) lens[tid] = lengths[tid];
  __syncthreads();

  const __bf16* whh_ld = whh + (size_t)(layer * 2 + dir) * 1024 * 256;  // [4H][H] row-major
  const int nbase = dir * 1024;
  const int mlane = (lane >> 4) << 3;    // +8 for upper lane half (C/D layout)
  const int collane = lane & 15;

  int m0[2], h0[2];
  float creg[2][8];
#pragma unroll
  for (int g = 0; g < 2; ++g) {
    int id = wave * 2 + g;               // 64 groups: 4 m-tiles x 16 h-tiles
    m0[g] = (id & 3) * 16;
    h0[g] = (id >> 2) * 16;
#pragma unroll
    for (int r = 0; r < 8; ++r) creg[g][r] = 0.0f;
  }

  for (int t = 0; t < 512; ++t) {
    float hv[2][8];
#pragma unroll
    for (int g = 0; g < 2; ++g) {
      const int colh = h0[g] + collane;
      v8f acc[4];
      // C init = xg at (b, u) with u = rev_idx for backward direction
#pragma unroll
      for (int gate = 0; gate < 4; ++gate) {
#pragma unroll
        for (int r = 0; r < 8; ++r) {
          int b = m0[g] + mlane + r;
          int len = lens[b];
          int u = (dir == 0) ? t : ((t < len) ? (len - 1 - t) : t);
          acc[gate][r] = xg[((size_t)(b * 512 + u)) * 2048 + nbase + gate * 256 + colh];
        }
      }
      // h @ w_hh^T over K = 256
#pragma unroll
      for (int kt = 0; kt < 8; ++kt) {
        const int k0 = kt * 32;
        v16bf a = load_a16x32(&h_lds[m0[g]][k0], 256, lane);
#pragma unroll
        for (int gate = 0; gate < 4; ++gate) {
          const __bf16* wb = whh_ld + (size_t)(gate * 256 + h0[g]) * 256 + k0;
          v16bf bm = load_b32x16(wb, 256, lane);
          acc[gate] = wmma_bf16(a, bm, acc[gate]);
        }
      }
      // gates: i,f,g,o ; c/h update in registers
#pragma unroll
      for (int r = 0; r < 8; ++r) {
        float iv = sigf(acc[0][r]);
        float fv = sigf(acc[1][r]);
        float gv = tanhf(acc[2][r]);
        float ov = sigf(acc[3][r]);
        creg[g][r] = fv * creg[g][r] + iv * gv;
        hv[g][r]   = ov * tanhf(creg[g][r]);
      }
    }
    __syncthreads();   // all reads of h(t-1) complete
#pragma unroll
    for (int g = 0; g < 2; ++g) {
      const int colh = h0[g] + collane;
#pragma unroll
      for (int r = 0; r < 8; ++r) {
        int b = m0[g] + mlane + r;
        int len = lens[b];
        h_lds[b][colh] = (__bf16)hv[g][r];
        int t_out = (dir == 0) ? t : ((t < len) ? (len - 1 - t) : t);
        float hm = (t < len) ? hv[g][r] : 0.0f;   // mask(b, t_out) == (t < len)
        out[((size_t)(b * 512 + t_out)) * 512 + dir * 256 + colh] = (__bf16)hm;
      }
    }
    __syncthreads();   // h(t) visible for next step
  }
}

// ---------------------------------------------------------------------------
// emissions[m][tag] = x2[m] . fc_w[:,tag] + fc_b[tag]   (N=20, VALU path)
// ---------------------------------------------------------------------------
__global__ void __launch_bounds__(256) emissions_kernel(const __bf16* __restrict__ X,
                                                        const float* __restrict__ fcw,
                                                        const float* __restrict__ fcb,
                                                        float* __restrict__ em) {
  const int tag = threadIdx.x & 31;
  const int m = blockIdx.x * 8 + (threadIdx.x >> 5);
  if (tag >= 20) return;
  float acc = fcb[tag];
  const __bf16* xr = X + (size_t)m * 512;
  for (int k = 0; k < 512; ++k) acc += (float)xr[k] * fcw[k * 20 + tag];
  em[(size_t)m * 20 + tag] = acc;
}

// ---------------------------------------------------------------------------
// CRF: one wave per batch row; states in lanes 0..19.
// ---------------------------------------------------------------------------
__global__ void __launch_bounds__(32) crf_kernel(const int* __restrict__ lengths,
                                                 const int* __restrict__ tags,
                                                 const float* __restrict__ em,
                                                 const float* __restrict__ cs,
                                                 const float* __restrict__ ce,
                                                 const float* __restrict__ ct,
                                                 float* __restrict__ out) {
  const int b = blockIdx.x, tid = threadIdx.x;
  __shared__ float trans[20][20];
  __shared__ float sc[20];
  for (int i = tid; i < 400; i += 32) trans[i / 20][i % 20] = ct[i];
  const int len = lengths[b];
  __syncthreads();

  // numerator (gold path score), strided over t
  float part = 0.0f;
  for (int t = tid; t < 512; t += 32) {
    if (t < len) {
      int tg = tags[b * 512 + t];
      part += em[(size_t)(b * 512 + t) * 20 + tg];
      if (t >= 1) part += trans[tags[b * 512 + t - 1]][tg];
    }
  }
#pragma unroll
  for (int off = 16; off > 0; off >>= 1) part += __shfl_down(part, off, 32);

  // forward algorithm (log-partition)
  float s = -1e30f;
  if (tid < 20) s = cs[tid] + em[(size_t)(b * 512) * 20 + tid];
  for (int t = 1; t < 512; ++t) {
    if (t >= len) break;                       // mask is monotone in t
    if (tid < 20) sc[tid] = s;
    __syncthreads();
    if (tid < 20) {
      float mx = -1e30f;
#pragma unroll
      for (int i = 0; i < 20; ++i) mx = fmaxf(mx, sc[i] + trans[i][tid]);
      float sum = 0.0f;
#pragma unroll
      for (int i = 0; i < 20; ++i) sum += __expf(sc[i] + trans[i][tid] - mx);
      s = mx + __logf(sum) + em[(size_t)(b * 512 + t) * 20 + tid];
    }
    __syncthreads();
  }
  float v = (tid < 20) ? s + ce[tid] : -1e30f;
  float mx = v;
#pragma unroll
  for (int off = 16; off > 0; off >>= 1) mx = fmaxf(mx, __shfl_xor(mx, off, 32));
  float ex = (tid < 20) ? __expf(v - mx) : 0.0f;
#pragma unroll
  for (int off = 16; off > 0; off >>= 1) ex += __shfl_xor(ex, off, 32);
  if (tid == 0) {
    float num = part + cs[tags[b * 512]] + ce[tags[b * 512 + len - 1]];
    float lse = mx + __logf(ex);
    atomicAdd(out, -(num - lse));
  }
}

// ---------------------------------------------------------------------------
extern "C" void kernel_launch(void* const* d_in, const int* in_sizes, int n_in,
                              void* d_out, int out_size, void* d_ws, size_t ws_size,
                              hipStream_t stream) {
  const int*   ids   = (const int*)d_in[0];
  const int*   lens  = (const int*)d_in[1];
  const int*   tags  = (const int*)d_in[2];
  const float* embed = (const float*)d_in[3];
  const float* w_ih  = (const float*)d_in[4];   // [L][2][4H][E]
  const float* w_hh  = (const float*)d_in[5];   // [L][2][4H][H]
  const float* b_ih  = (const float*)d_in[6];
  const float* b_hh  = (const float*)d_in[7];
  const float* fc_w  = (const float*)d_in[8];
  const float* fc_b  = (const float*)d_in[9];
  const float* c_s   = (const float*)d_in[10];
  const float* c_e   = (const float*)d_in[11];
  const float* c_t   = (const float*)d_in[12];
  float* outp = (float*)d_out;

  char* ws = (char*)d_ws;
  __bf16* bufA = (__bf16*)(ws);                      //  32768*512 bf16 = 32 MB
  __bf16* bufB = (__bf16*)(ws + 33554432);           //  32 MB
  float*  xg   = (float*)(ws + 67108864);            //  32768*2048 f32 = 256 MB
  __bf16* wihb = (__bf16*)(ws + 335544320);          //  4 MB
  __bf16* whhb = (__bf16*)(ws + 339738624);          //  2 MB
  float*  em   = (float*)(ws + 341835776);           //  2.5 MB

  zero_kernel<<<1, 32, 0, stream>>>(outp);
  f32_to_bf16_kernel<<<(2097152 + 255) / 256, 256, 0, stream>>>(w_ih, wihb, 2097152);
  f32_to_bf16_kernel<<<(1048576 + 255) / 256, 256, 0, stream>>>(w_hh, whhb, 1048576);
  gather_embed_kernel<<<8192, 256, 0, stream>>>(ids, embed, bufA);

  // layer 0
  xg_gemm_kernel<<<dim3(256, 32), 256, 0, stream>>>(bufA, wihb, b_ih, b_hh, xg, 0);
  lstm_scan_kernel<<<2, 1024, 0, stream>>>(xg, whhb, lens, bufB, 0);
  // layer 1
  xg_gemm_kernel<<<dim3(256, 32), 256, 0, stream>>>(bufB, wihb, b_ih, b_hh, xg, 1);
  lstm_scan_kernel<<<2, 1024, 0, stream>>>(xg, whhb, lens, bufA, 1);

  emissions_kernel<<<4096, 256, 0, stream>>>(bufA, fc_w, fc_b, em);
  crf_kernel<<<64, 32, 0, stream>>>(lens, tags, em, c_s, c_e, c_t, outp);
}
